// MLPRegressor_76072460746998
// MI455X (gfx1250) — matrix operand order repaired and checked by reading, not compiled, not run on hardware
//
#include <hip/hip_runtime.h>
#include <hip/hip_bf16.h>

typedef __attribute__((ext_vector_type(16))) _Float16 v16h;
typedef __attribute__((ext_vector_type(8)))  _Float16 v8h;
typedef __attribute__((ext_vector_type(8)))  float    v8f;
typedef __attribute__((ext_vector_type(4)))  float    v4f;

#define NB   2048
#define NS   512
#define EMB  64
// CARD = {2,2,2,11,19,31,34}; prefix offsets {0,2,4,6,17,36,67}; total rows = 101

#define AS1 __attribute__((address_space(1)))
#define AS3 __attribute__((address_space(3)))

// ---- CDNA5 async global->LDS staging (ASYNCcnt path), feature-guarded -----
#if __has_builtin(__builtin_amdgcn_global_load_async_to_lds_b32) && \
    __has_builtin(__builtin_amdgcn_s_wait_asynccnt)
#define HAVE_ASYNC_LDS 1
#else
#define HAVE_ASYNC_LDS 0
#endif

__device__ __forceinline__ void g2l_f32(float* lds_dst, const float* gsrc) {
#if HAVE_ASYNC_LDS
    __builtin_amdgcn_global_load_async_to_lds_b32((AS1 int*)gsrc,
                                                  (AS3 int*)lds_dst, 0, 0);
#else
    *lds_dst = *gsrc;
#endif
}

__device__ __forceinline__ void g2l_i32(int* lds_dst, const int* gsrc) {
#if HAVE_ASYNC_LDS
    __builtin_amdgcn_global_load_async_to_lds_b32((AS1 int*)gsrc,
                                                  (AS3 int*)lds_dst, 0, 0);
#else
    *lds_dst = *gsrc;
#endif
}

__device__ __forceinline__ void async_wait_all() {
#if HAVE_ASYNC_LDS
    __builtin_amdgcn_s_wait_asynccnt(0);
#endif
}

// ---------------------------------------------------------------------------
// Kernel 1: masked-mean pooling of [cat-embedding-mean || relu(cont @ Wc + bc)]
// One block per batch row. Tables + W_cont cached in LDS; cat_x / cont_x
// streamed through LDS in 64-position chunks via async-to-LDS, only while
// s < len. Bandwidth-bound: ~48MB worst case -> ~2us at 23.3 TB/s.
// ---------------------------------------------------------------------------
__global__ __launch_bounds__(128) void pool_kernel(
    const float* __restrict__ cont_x,  // [B,S,5]
    const int*   __restrict__ cat_x,   // [B,S,7]
    const int*   __restrict__ length,  // [B]
    const float* __restrict__ e0, const float* __restrict__ e1,
    const float* __restrict__ e2, const float* __restrict__ e3,
    const float* __restrict__ e4, const float* __restrict__ e5,
    const float* __restrict__ e6,
    const float* __restrict__ W_cont,  // [5,64]
    const float* __restrict__ b_cont,  // [64]
    float* __restrict__ pooled)        // [B,128]
{
    __shared__ float sT[101 * EMB];    // concatenated embedding tables
    __shared__ float sW[5 * EMB];
    __shared__ float sB[EMB];
    __shared__ float sCont[64 * 5];
    __shared__ int   sCat[64 * 7];

    const float* tabs[7] = {e0, e1, e2, e3, e4, e5, e6};
    const int    offs[7] = {0, 2, 4, 6, 17, 36, 67};
    const int    card[7] = {2, 2, 2, 11, 19, 31, 34};

    for (int i = 0; i < 7; ++i) {
        const int n = card[i] * EMB;
        for (int t = threadIdx.x; t < n; t += 128) g2l_f32(&sT[offs[i] * EMB + t], &tabs[i][t]);
    }
    for (int t = threadIdx.x; t < 5 * EMB; t += 128) g2l_f32(&sW[t], &W_cont[t]);
    if (threadIdx.x < EMB) g2l_f32(&sB[threadIdx.x], &b_cont[threadIdx.x]);
    async_wait_all();
    __syncthreads();

    const int b   = blockIdx.x;
    const int len = length[b];           // >= 1 by construction
    const int d   = threadIdx.x;         // feature id 0..127

    const float* cx = cont_x + (size_t)b * NS * 5;
    const int*   ct = cat_x  + (size_t)b * NS * 7;

    float acc = 0.0f;
    for (int s0 = 0; s0 < len; s0 += 64) {
        const int ns = min(64, len - s0);
        for (int t = threadIdx.x; t < ns * 5; t += 128) g2l_f32(&sCont[t], &cx[s0 * 5 + t]);
        for (int t = threadIdx.x; t < ns * 7; t += 128) g2l_i32(&sCat[t],  &ct[s0 * 7 + t]);
        async_wait_all();
        __syncthreads();

        // prefetch next chunk (speculative; one 128B line per participating lane)
        if (s0 + 64 < len && s0 + 128 <= NS) {
            const char* pc = (const char*)&cx[(s0 + 64) * 5];   // 1280 B -> 10 lines
            const char* pt = (const char*)&ct[(s0 + 64) * 7];   // 1792 B -> 14 lines
            if (threadIdx.x < 10)      __builtin_prefetch(pc + threadIdx.x * 128, 0, 1);
            else if (threadIdx.x < 24) __builtin_prefetch(pt + (threadIdx.x - 10) * 128, 0, 1);
        }

        if (d < EMB) {
            for (int s = 0; s < ns; ++s) {
                const int* c = &sCat[s * 7];
                acc += sT[(0  + c[0]) * EMB + d] + sT[(2  + c[1]) * EMB + d]
                     + sT[(4  + c[2]) * EMB + d] + sT[(6  + c[3]) * EMB + d]
                     + sT[(17 + c[4]) * EMB + d] + sT[(36 + c[5]) * EMB + d]
                     + sT[(67 + c[6]) * EMB + d];
            }
        } else {
            const int j = d - EMB;
            for (int s = 0; s < ns; ++s) {
                const float* q = &sCont[s * 5];
                float v = sB[j];
                v = fmaf(q[0], sW[0 * EMB + j], v);
                v = fmaf(q[1], sW[1 * EMB + j], v);
                v = fmaf(q[2], sW[2 * EMB + j], v);
                v = fmaf(q[3], sW[3 * EMB + j], v);
                v = fmaf(q[4], sW[4 * EMB + j], v);
                acc += fmaxf(v, 0.0f);
            }
        }
        __syncthreads();
    }

    const float inv = 1.0f / (float)len;
    pooled[(size_t)b * 128 + d] = (d < EMB) ? acc * inv * (1.0f / 7.0f) : acc * inv;
}

// ---------------------------------------------------------------------------
// Kernel 2: MLP head with WMMA (f16 inputs, f32 accumulate).
//   h   = relu(pooled[2048,128] @ W1[128,64] + b1)   -> 16 WMMAs / wave
//   out = relu(h @ W2pad[64,16] + b2)[:, :2]         ->  2 WMMAs / wave
// Block = 128 threads = 4 waves; each wave owns a 16-row tile.
// Weights are stored TRANSPOSED + padded in LDS so every B-fragment is one
// contiguous 32B v16h read (2x ds_load_b128, bank-rotated strides).
// A/B fragments use contiguous K = 16g + j order in both operands, not the ISA table order (wave32).
// ---------------------------------------------------------------------------
#define W1T_STRIDE 144   // 288B rows: 32B-aligned vectors, 8-bank rotation/lane
#define W2T_STRIDE 80    // 160B rows: 32B-aligned vectors, 40-bank rotation/lane
#define H_STRIDE   72    // 144B rows: 16B-aligned vectors, 36-bank rotation/lane

__global__ __launch_bounds__(128) void head_kernel(
    const float* __restrict__ pooled,  // [2048,128]
    const float* __restrict__ W1,      // [128,64]
    const float* __restrict__ b1,      // [64]
    const float* __restrict__ W2,      // [64,2]
    const float* __restrict__ b2,      // [2]
    float* __restrict__ out)           // [2048,2]
{
    __shared__ _Float16 sW1t[64 * W1T_STRIDE];   // [col][K]
    __shared__ _Float16 sW2t[16 * W2T_STRIDE];   // [col][K], cols 2..15 zero
    __shared__ _Float16 sH[4][16 * H_STRIDE];    // per-wave h tile [M][K]

    // stage W1 transposed: sW1t[c][k] = W1[k][c]; contiguous LDS writes
    for (int t = threadIdx.x; t < 128 * 64; t += 128) {
        const int k = t & 127, c = t >> 7;
        sW1t[c * W1T_STRIDE + k] = (_Float16)W1[k * 64 + c];
    }
    // stage W2 transposed + zero-padded to 16 cols
    for (int t = threadIdx.x; t < 16 * 64; t += 128) {
        const int k = t & 63, c = t >> 6;
        sW2t[c * W2T_STRIDE + k] = (_Float16)((c < 2) ? W2[k * 2 + c] : 0.0f);
    }
    __syncthreads();

    const int wave = threadIdx.x >> 5;
    const int lane = threadIdx.x & 31;
    const int g    = lane >> 4;        // half-wave group
    const int ln   = lane & 15;
    const int row0 = (blockIdx.x * 4 + wave) * 16;

    // ---- GEMM1: pooled(16x128) @ W1(128x64) -> 4 N-tiles of 16x16 f32 ----
    v8f acc[4] = {{}, {}, {}, {}};
    const float* Arow = pooled + (size_t)row0 * 128;
    for (int ks = 0; ks < 4; ++ks) {
        // A fragment: element j -> K = ks*32 + 16g + j
        const float* ap = Arow + (size_t)ln * 128 + ks * 32 + 16 * g;
        const v4f r0 = *(const v4f*)(ap);
        const v4f r1 = *(const v4f*)(ap + 4);
        const v4f r2 = *(const v4f*)(ap + 8);
        const v4f r3 = *(const v4f*)(ap + 12);
        v16h a;
#pragma unroll
        for (int i = 0; i < 4; ++i) {
            a[i]      = (_Float16)r0[i];
            a[4 + i]  = (_Float16)r1[i];
            a[8 + i]  = (_Float16)r2[i];
            a[12 + i] = (_Float16)r3[i];
        }
#pragma unroll
        for (int n = 0; n < 4; ++n) {
            // B fragment: element j -> K = ks*32 + 16g + j (contiguous 32B)
            const v16h bf = *(const v16h*)&sW1t[(n * 16 + ln) * W1T_STRIDE + ks * 32 + 16 * g];
            acc[n] = __builtin_amdgcn_wmma_f32_16x16x32_f16(
                false, a, false, bf, (short)0, acc[n], false, false);
        }
    }

    // bias + relu, park h tile in LDS (C-layout -> row-major) for re-fragmenting
#pragma unroll
    for (int n = 0; n < 4; ++n) {
        const float bv = b1[n * 16 + ln];
#pragma unroll
        for (int r = 0; r < 8; ++r) {
            const int M = r + 8 * g;
            sH[wave][M * H_STRIDE + n * 16 + ln] = (_Float16)fmaxf(acc[n][r] + bv, 0.0f);
        }
    }
    __syncthreads();

    // ---- GEMM2: h(16x64) @ W2pad(64x16) ----
    v8f acc2 = {};
#pragma unroll
    for (int ks = 0; ks < 2; ++ks) {
        const _Float16* hp = &sH[wave][ln * H_STRIDE + ks * 32 + 16 * g];
        const v8h h0 = *(const v8h*)(hp);
        const v8h h1 = *(const v8h*)(hp + 8);
        v16h a;
#pragma unroll
        for (int i = 0; i < 8; ++i) { a[i] = h0[i]; a[8 + i] = h1[i]; }
        const v16h bf = *(const v16h*)&sW2t[ln * W2T_STRIDE + ks * 32 + 16 * g];
        acc2 = __builtin_amdgcn_wmma_f32_16x16x32_f16(
            false, a, false, bf, (short)0, acc2, false, false);
    }

    if (ln < 2) {
        const float bv = b2[ln];
#pragma unroll
        for (int r = 0; r < 8; ++r) {
            const int M = r + 8 * g;
            out[(size_t)(row0 + M) * 2 + ln] = fmaxf(acc2[r] + bv, 0.0f);
        }
    }
}

// ---------------------------------------------------------------------------
// Launcher. Inputs (setup_inputs order):
//  0 cont_x  1 cat_x  2 length  3..9 emb0..emb6
//  10 W_cont 11 b_cont 12 W1 13 b1 14 W2 15 b2
// ---------------------------------------------------------------------------
extern "C" void kernel_launch(void* const* d_in, const int* in_sizes, int n_in,
                              void* d_out, int out_size, void* d_ws, size_t ws_size,
                              hipStream_t stream) {
    const float* cont_x = (const float*)d_in[0];
    const int*   cat_x  = (const int*)  d_in[1];
    const int*   length = (const int*)  d_in[2];
    const float* e0 = (const float*)d_in[3];
    const float* e1 = (const float*)d_in[4];
    const float* e2 = (const float*)d_in[5];
    const float* e3 = (const float*)d_in[6];
    const float* e4 = (const float*)d_in[7];
    const float* e5 = (const float*)d_in[8];
    const float* e6 = (const float*)d_in[9];
    const float* W_cont = (const float*)d_in[10];
    const float* b_cont = (const float*)d_in[11];
    const float* W1 = (const float*)d_in[12];
    const float* b1 = (const float*)d_in[13];
    const float* W2 = (const float*)d_in[14];
    const float* b2 = (const float*)d_in[15];

    float* pooled = (float*)d_ws;              // [2048,128] f32 = 1 MB scratch
    float* outp   = (float*)d_out;             // [2048,2] f32

    pool_kernel<<<NB, 128, 0, stream>>>(cont_x, cat_x, length,
                                        e0, e1, e2, e3, e4, e5, e6,
                                        W_cont, b_cont, pooled);

    head_kernel<<<NB / 64, 128, 0, stream>>>(pooled, W1, b1, W2, b2, outp);
}
